// Cosal_Module_55525337203074
// MI455X (gfx1250) — matrix operand order repaired and checked by reading, not compiled, not run on hardware
//
#include <hip/hip_runtime.h>

typedef __bf16 bf16;
typedef __attribute__((ext_vector_type(16))) bf16 bfrag;
typedef __attribute__((ext_vector_type(8))) float ffrag;

#define NIMG 32
#define CCH  512
#define HW   1024
#define P34  34
#define P2   (34 * 34)

struct alignas(16) Q16 { unsigned int w[4]; };

__device__ inline bfrag load8x2(const bf16* p0, const bf16* p1) {
  bfrag f;
  *reinterpret_cast<Q16*>(&f)       = *reinterpret_cast<const Q16*>(p0);
  *(reinterpret_cast<Q16*>(&f) + 1) = *reinterpret_cast<const Q16*>(p1);
  return f;
}

// A fragment: 16x32 bf16 tile, row-major (row m, col k), lda elements.
// wave32 layout: lane = g*16+m ; elems 0..7 = K[8g,8g+8) ; elems 8..15 = K[16+8g,16+8g+8)
__device__ inline bfrag load_frag_a(const bf16* base, int lda) {
  int lane = threadIdx.x & 31;
  int g = lane >> 4, m = lane & 15;
  const bf16* r = base + (size_t)m * lda;
  return load8x2(r + 8 * g, r + 16 + 8 * g);
}

// B fragment from Bt (column-major B): Bt[col][k], ldbt elements.
// lane = g*16+n ; elems 0..7 = K[8g,8g+8) ; elems 8..15 = K[16+8g,16+8g+8)
__device__ inline bfrag load_frag_bt(const bf16* base, int ldbt) {
  int lane = threadIdx.x & 31;
  int g = lane >> 4, n = lane & 15;
  const bf16* r = base + (size_t)n * ldbt;
  return load8x2(r + 8 * g, r + 16 + 8 * g);
}

__device__ inline ffrag fzero() {
  ffrag z = {0.f, 0.f, 0.f, 0.f, 0.f, 0.f, 0.f, 0.f};
  return z;
}

__device__ inline ffrag wmma_bf16(bfrag a, bfrag b, ffrag c) {
  return __builtin_amdgcn_wmma_f32_16x16x32_bf16(false, a, false, b, (short)0, c,
                                                 false, false);
}

template <int NT>
__device__ inline void wmma_tiles(const bf16* A, int lda, const bf16* Bt, int ldbt,
                                  int K, ffrag* acc) {
  for (int k = 0; k < K; k += 32) {
    bfrag a = load_frag_a(A + k, lda);
#pragma unroll
    for (int t = 0; t < NT; ++t) {
      bfrag b = load_frag_bt(Bt + (size_t)t * 16 * ldbt + k, ldbt);
      acc[t] = wmma_bf16(a, b, acc[t]);
    }
  }
}

// C/D 16x16 f32 layout: lane = g*16+n ; elem j -> (row 8g+j, col n)
__device__ inline void store_tile_f32(float* out, int ldo, ffrag c) {
  int lane = threadIdx.x & 31, g = lane >> 4, n = lane & 15;
#pragma unroll
  for (int j = 0; j < 8; ++j) out[(size_t)(8 * g + j) * ldo + n] = c[j];
}

// ---------------- normalization / small kernels ----------------

// Per-pixel channel L2 norm -> NFs in two bf16 layouts:
//   nfsB[n][c][q]  (B operand for GEMM1)
//   nfsT[n][q][c]  (A for corr GEMM, Bt for GEMM2)
__global__ void k_norm_feats(const float* __restrict__ feats, bf16* __restrict__ nfsB,
                             bf16* __restrict__ nfsT) {
  int n = blockIdx.y;
  int q = blockIdx.x * blockDim.x + threadIdx.x;
  const float* f = feats + (size_t)n * CCH * HW + q;
  float s = 0.f;
  for (int c = 0; c < CCH; ++c) { float v = f[(size_t)c * HW]; s += v * v; }
  float inv = 1.f / fmaxf(sqrtf(s), 1e-12f);
  bf16* pB = nfsB + (size_t)n * CCH * HW + q;
  bf16* pT = nfsT + ((size_t)n * HW + q) * CCH;
  for (int c = 0; c < CCH; ++c) {
    float v = f[(size_t)c * HW] * inv;
    pB[(size_t)c * HW] = (bf16)v;
    pT[c] = (bf16)v;
  }
}

// Masked mean over pixels for both masks -> sivraw[mask][n][c]
__global__ void k_siv(const bf16* __restrict__ nfsB, const float* __restrict__ sisms,
                      float* __restrict__ sivraw) {
  int c = blockIdx.x, n = blockIdx.y, t = threadIdx.x;
  __shared__ float sf[128], sb[128];
  const bf16* src = nfsB + ((size_t)n * CCH + c) * HW;
  const float* msk = sisms + (size_t)n * HW;
  float af = 0.f, ab = 0.f;
  for (int i = t; i < HW; i += 128) {
    float v = (float)src[i], s = msk[i];
    af += v * s;
    ab += v * (1.f - s);
  }
  sf[t] = af; sb[t] = ab;
  __syncthreads();
  for (int o = 64; o > 0; o >>= 1) {
    if (t < o) { sf[t] += sf[t + o]; sb[t] += sb[t + o]; }
    __syncthreads();
  }
  if (t == 0) {
    sivraw[(size_t)n * CCH + c] = sf[0] / (float)HW;
    sivraw[((size_t)NIMG + n) * CCH + c] = sb[0] / (float)HW;
  }
}

// L2-normalize SIVs over channels -> bf16 [mask][n][c] (row-major = Bt layout)
__global__ void k_siv_norm(const float* __restrict__ sivraw, bf16* __restrict__ sivbf) {
  int n = blockIdx.x, mask = blockIdx.y, t = threadIdx.x;
  const float* src = sivraw + ((size_t)mask * NIMG + n) * CCH;
  bf16* dst = sivbf + ((size_t)mask * NIMG + n) * CCH;
  __shared__ float red[256];
  float s = 0.f;
  for (int c = t; c < CCH; c += 256) { float v = src[c]; s += v * v; }
  red[t] = s;
  __syncthreads();
  for (int o = 128; o > 0; o >>= 1) {
    if (t < o) red[t] += red[t + o];
    __syncthreads();
  }
  float inv = 1.f / fmaxf(sqrtf(red[0]), 1e-12f);
  for (int c = t; c < CCH; c += 256) dst[c] = (bf16)(src[c] * inv);
}

// corr[n][p][m] = sum_c NFs[n,c,p] * SIV[m,c]   (WMMA: M=1024 rows p, N=32 cols m, K=512)
__global__ void k_gemm_corr(const bf16* __restrict__ nfsT, const bf16* __restrict__ sivm,
                            float* __restrict__ corr) {
  int n = blockIdx.y;
  int wave = threadIdx.x >> 5;
  int p0 = blockIdx.x * 64 + wave * 16;
  const bf16* A = nfsT + ((size_t)n * HW + p0) * CCH;
  ffrag acc[2] = {fzero(), fzero()};
  wmma_tiles<2>(A, CCH, sivm, CCH, CCH, acc);
  float* out = corr + ((size_t)n * HW + p0) * 32;
  store_tile_f32(out, 32, acc[0]);
  store_tile_f32(out + 16, 32, acc[1]);
}

// l2norm of corr over p (axis=HW) per (n,m), in place
__global__ void k_cm_norm(float* __restrict__ corr) {
  int m = blockIdx.x, n = blockIdx.y, t = threadIdx.x;
  float* base = corr + (size_t)n * HW * 32 + m;
  __shared__ float red[256];
  float s = 0.f;
  for (int p = t; p < HW; p += 256) { float v = base[(size_t)p * 32]; s += v * v; }
  red[t] = s;
  __syncthreads();
  for (int o = 128; o > 0; o >>= 1) {
    if (t < o) red[t] += red[t + o];
    __syncthreads();
  }
  float inv = 1.f / fmaxf(sqrtf(red[0]), 1e-12f);
  for (int p = t; p < HW; p += 256) base[(size_t)p * 32] *= inv;
}

// wv softmax + CSA + minmax normalization, one block per image.
// Uses identity sum_q corr_mat[n,p,q] = sum_k cm[n,p,k] * (sum_q cm[n,q,k])
__global__ void k_csa(const float* __restrict__ corr, float* __restrict__ csa) {
  int n = blockIdx.x, t = threadIdx.x;
  __shared__ float tk[HW];
  __shared__ float red[256];
  __shared__ float Sv[32];
  __shared__ float wv[32];
  __shared__ float mm[2];
  const float* cmn = corr + (size_t)n * HW * 32;
  for (int k = t; k < HW; k += 256) {
    const float* r = cmn + (size_t)k * 32;
    float s = 0.f;
    for (int m = 0; m < 32; ++m) s += r[m];
    tk[k] = s;
  }
  __syncthreads();
  float part[32];
  for (int m = 0; m < 32; ++m) part[m] = 0.f;
  for (int k = t; k < HW; k += 256) {
    float w = tk[k];
    const float* r = cmn + (size_t)k * 32;
    for (int m = 0; m < 32; ++m) part[m] += r[m] * w;
  }
  for (int m = 0; m < 32; ++m) {
    red[t] = part[m];
    __syncthreads();
    for (int o = 128; o > 0; o >>= 1) {
      if (t < o) red[t] += red[t + o];
      __syncthreads();
    }
    if (t == 0) Sv[m] = red[0];
    __syncthreads();
  }
  if (t == 0) {
    float mx = Sv[0];
    for (int m = 1; m < 32; ++m) mx = fmaxf(mx, Sv[m]);
    float sum = 0.f;
    for (int m = 0; m < 32; ++m) { float e = __expf(Sv[m] - mx); wv[m] = e; sum += e; }
    for (int m = 0; m < 32; ++m) wv[m] /= sum;
  }
  __syncthreads();
  float lmn = 1e30f, lmx = -1e30f;
  for (int k = t; k < HW; k += 256) {
    const float* r = cmn + (size_t)k * 32;
    float s = 0.f;
    for (int m = 0; m < 32; ++m) s += r[m] * wv[m];
    tk[k] = s;
    lmn = fminf(lmn, s);
    lmx = fmaxf(lmx, s);
  }
  red[t] = lmn; __syncthreads();
  for (int o = 128; o > 0; o >>= 1) { if (t < o) red[t] = fminf(red[t], red[t + o]); __syncthreads(); }
  if (t == 0) mm[0] = red[0];
  __syncthreads();
  red[t] = lmx; __syncthreads();
  for (int o = 128; o > 0; o >>= 1) { if (t < o) red[t] = fmaxf(red[t], red[t + o]); __syncthreads(); }
  if (t == 0) mm[1] = red[0];
  __syncthreads();
  float mn = mm[0], rng = mm[1] - mm[0] + 1e-12f;
  for (int k = t; k < HW; k += 256) csa[(size_t)n * HW + k] = (tk[k] - mn) / rng;
}

// rank[p] = position of p in stable descending argsort of CSA (O(HW^2), trivial)
__global__ void k_rank(const float* __restrict__ csa, int* __restrict__ rank) {
  int n = blockIdx.x, t = threadIdx.x;
  __shared__ float v[HW];
  const float* src = csa + (size_t)n * HW;
  for (int k = t; k < HW; k += 256) v[k] = src[k];
  __syncthreads();
  for (int p = t; p < HW; p += 256) {
    float vp = v[p];
    int r = 0;
    for (int j = 0; j < HW; ++j) {
      float vj = v[j];
      r += (vj > vp) || ((vj == vp) && (j < p));
    }
    rank[(size_t)n * HW + p] = r;
  }
}

// cwp[n][o][p] = cw[o][rank[n][p]]  (bf16)
__global__ void k_permw(const float* __restrict__ cw, const int* __restrict__ rank,
                        bf16* __restrict__ cwp) {
  int o = blockIdx.x, n = blockIdx.y, t = threadIdx.x;
  const float* wrow = cw + (size_t)o * HW;
  const int* rk = rank + (size_t)n * HW;
  bf16* dst = cwp + ((size_t)n * 128 + o) * HW;
  for (int p = t; p < HW; p += 256) dst[p] = (bf16)wrow[rk[p]];
}

// P[n][o][c] = sum_p cwp[n][o][p] * NFs[n][c][p]   (M=128, N=512, K=1024)
__global__ void k_gemm1(const bf16* __restrict__ cwp, const bf16* __restrict__ nfsB,
                        bf16* __restrict__ pbf) {
  int n = blockIdx.z;
  int m0 = blockIdx.x * 16;
  int wave = threadIdx.x >> 5;
  int c0 = blockIdx.y * 256 + wave * 64;
  const bf16* A = cwp + ((size_t)n * 128 + m0) * HW;
  const bf16* Bt = nfsB + ((size_t)n * CCH + c0) * HW;
  ffrag acc[4] = {fzero(), fzero(), fzero(), fzero()};
  wmma_tiles<4>(A, HW, Bt, HW, HW, acc);
  int lane = threadIdx.x & 31, g = lane >> 4, nn = lane & 15;
#pragma unroll
  for (int tt = 0; tt < 4; ++tt) {
    bf16* out = pbf + ((size_t)n * 128 + m0) * CCH + c0 + tt * 16;
#pragma unroll
    for (int j = 0; j < 8; ++j) out[(size_t)(8 * g + j) * CCH + nn] = (bf16)acc[tt][j];
  }
}

// x[n][o][q] = CSA[n][q] * (sum_c P[n][o][c] * NFs[n][c][q]) + cb[o]
__global__ void k_gemm2(const bf16* __restrict__ pbf, const bf16* __restrict__ nfsT,
                        const float* __restrict__ csa, const float* __restrict__ cb,
                        float* __restrict__ x) {
  int n = blockIdx.z;
  int m0 = blockIdx.x * 16;
  int wave = threadIdx.x >> 5;
  int q0 = blockIdx.y * 256 + wave * 64;
  const bf16* A = pbf + ((size_t)n * 128 + m0) * CCH;
  const bf16* Bt = nfsT + ((size_t)n * HW + q0) * CCH;
  ffrag acc[4] = {fzero(), fzero(), fzero(), fzero()};
  wmma_tiles<4>(A, CCH, Bt, CCH, CCH, acc);
  int lane = threadIdx.x & 31, g = lane >> 4, nn = lane & 15;
#pragma unroll
  for (int tt = 0; tt < 4; ++tt) {
#pragma unroll
    for (int j = 0; j < 8; ++j) {
      int row = m0 + 8 * g + j;
      int q = q0 + tt * 16 + nn;
      x[((size_t)n * 128 + row) * HW + q] = acc[tt][j] * csa[(size_t)n * HW + q] + cb[row];
    }
  }
}

// pad (optionally BN+relu) into pixel-major bf16 [n][34*34][128]
__global__ void k_pad(const float* __restrict__ src, bf16* __restrict__ padT,
                      const float* __restrict__ scale, const float* __restrict__ shift,
                      int use_bn) {
  int ps = blockIdx.x, n = blockIdx.y, c = threadIdx.x;
  int hp = ps / P34, wp = ps % P34;
  bf16* dst = padT + ((size_t)n * P2 + ps) * 128 + c;
  if (hp == 0 || hp == 33 || wp == 0 || wp == 33) {
    *dst = (bf16)0.f;
    return;
  }
  float v = src[((size_t)n * 128 + c) * HW + (hp - 1) * 32 + (wp - 1)];
  if (use_bn) v = fmaxf(v * scale[c] + shift[c], 0.f);
  *dst = (bf16)v;
}

// 3x3 conv as 9 shifted 128x128 WMMA GEMMs over pixel-major padded input.
// mode 0: out = acc + bias (to outf)
// mode 1: out = relu(resid + acc + bias) (to outf, f32 [n][o][q])
// mode 2: out = relu(resid + acc + bias) (to outb, bf16 catT [n][q][coff+o])
__global__ void k_conv3(const bf16* __restrict__ wpk, const bf16* __restrict__ padT,
                        const float* __restrict__ bias, const float* __restrict__ resid,
                        float* __restrict__ outf, bf16* __restrict__ outb, int mode,
                        int coff) {
  int n = blockIdx.y;
  int h = blockIdx.x >> 1;
  int w0 = (blockIdx.x & 1) * 16;
  int o0 = (threadIdx.x >> 5) * 64;
  const bf16* ptn = padT + (size_t)n * P2 * 128;
  ffrag acc[4] = {fzero(), fzero(), fzero(), fzero()};
  for (int s = 0; s < 9; ++s) {
    int dh = s / 3, dw = s % 3;
    const bf16* btb = ptn + ((size_t)(h + dh) * P34 + (w0 + dw)) * 128;
#pragma unroll
    for (int kk = 0; kk < 4; ++kk) {
      bfrag b = load_frag_bt(btb + kk * 32, 128);
#pragma unroll
      for (int tt = 0; tt < 4; ++tt) {
        bfrag a = load_frag_a(wpk + (size_t)s * 16384 + (size_t)(o0 + tt * 16) * 128 + kk * 32, 128);
        acc[tt] = wmma_bf16(a, b, acc[tt]);
      }
    }
  }
  int lane = threadIdx.x & 31, g = lane >> 4, nn = lane & 15;
  int q = h * 32 + w0 + nn;
#pragma unroll
  for (int tt = 0; tt < 4; ++tt) {
#pragma unroll
    for (int j = 0; j < 8; ++j) {
      int row = o0 + tt * 16 + 8 * g + j;
      float v = acc[tt][j] + bias[row];
      if (mode == 0) {
        outf[((size_t)n * 128 + row) * HW + q] = v;
      } else {
        v = fmaxf(resid[((size_t)n * 128 + row) * HW + q] + v, 0.f);
        if (mode == 1)
          outf[((size_t)n * 128 + row) * HW + q] = v;
        else
          outb[((size_t)n * HW + q) * 256 + coff + row] = (bf16)v;
      }
    }
  }
}

// training-mode BN batch stats -> fused scale/shift
__global__ void k_bn_stats(const float* __restrict__ z, const float* __restrict__ gamma,
                           const float* __restrict__ beta, float* __restrict__ scale,
                           float* __restrict__ shift) {
  int o = blockIdx.x, t = threadIdx.x;
  __shared__ float rs[256], rq[256];
  float s = 0.f, sq = 0.f;
  for (int i = t; i < NIMG * HW; i += 256) {
    int n = i >> 10, q = i & 1023;
    float v = z[((size_t)n * 128 + o) * HW + q];
    s += v;
    sq += v * v;
  }
  rs[t] = s; rq[t] = sq;
  __syncthreads();
  for (int w = 128; w > 0; w >>= 1) {
    if (t < w) { rs[t] += rs[t + w]; rq[t] += rq[t + w]; }
    __syncthreads();
  }
  if (t == 0) {
    float inv_n = 1.f / (float)(NIMG * HW);
    float mean = rs[0] * inv_n;
    float var = rq[0] * inv_n - mean * mean;
    float sc = gamma[o] * rsqrtf(var + 1e-5f);
    scale[o] = sc;
    shift[o] = beta[o] - mean * sc;
  }
}

// outer 1x1 conv 256->128: x[n][o][q] = sum_i ow[o][i]*cat[n][i][q] + ob[o]
__global__ void k_gemm_out1x1(const bf16* __restrict__ owbf, const bf16* __restrict__ catT,
                              const float* __restrict__ ob, float* __restrict__ x) {
  int n = blockIdx.z;
  int m0 = blockIdx.x * 16;
  int wave = threadIdx.x >> 5;
  int q0 = blockIdx.y * 256 + wave * 64;
  const bf16* A = owbf + (size_t)m0 * 256;
  const bf16* Bt = catT + ((size_t)n * HW + q0) * 256;
  ffrag acc[4] = {fzero(), fzero(), fzero(), fzero()};
  wmma_tiles<4>(A, 256, Bt, 256, 256, acc);
  int lane = threadIdx.x & 31, g = lane >> 4, nn = lane & 15;
#pragma unroll
  for (int tt = 0; tt < 4; ++tt) {
#pragma unroll
    for (int j = 0; j < 8; ++j) {
      int row = m0 + 8 * g + j;
      int q = q0 + tt * 16 + nn;
      x[((size_t)n * 128 + row) * HW + q] = acc[tt][j] + ob[row];
    }
  }
}

// pack 3x3 conv weights [o][i][kh][kw] f32 -> wpk[s=kh*3+kw][o][i] bf16
__global__ void k_pack_convw(const float* __restrict__ w, bf16* __restrict__ wpk) {
  int idx = blockIdx.x * 256 + threadIdx.x;
  if (idx >= 9 * 128 * 128) return;
  int s = idx / 16384, rem = idx % 16384, o = rem / 128, i = rem % 128;
  int kh = s / 3, kw = s % 3;
  wpk[idx] = (bf16)w[(((size_t)o * 128 + i) * 3 + kh) * 3 + kw];
}

__global__ void k_pack_ow(const float* __restrict__ w, bf16* __restrict__ dst) {
  int idx = blockIdx.x * 256 + threadIdx.x;
  if (idx < 128 * 256) dst[idx] = (bf16)w[idx];
}

// ---------------- host orchestration ----------------

extern "C" void kernel_launch(void* const* d_in, const int* in_sizes, int n_in,
                              void* d_out, int out_size, void* d_ws, size_t ws_size,
                              hipStream_t stream) {
  (void)in_sizes; (void)n_in; (void)out_size; (void)ws_size;
  const float* feats   = (const float*)d_in[0];
  const float* sisms   = (const float*)d_in[1];
  const float* sub_cw  = (const float*)d_in[2];
  const float* sub_cb  = (const float*)d_in[3];
  const float* sub_rw1 = (const float*)d_in[4];
  const float* sub_rb1 = (const float*)d_in[5];
  const float* sub_rg  = (const float*)d_in[6];
  const float* sub_rbt = (const float*)d_in[7];
  const float* sub_rw2 = (const float*)d_in[8];
  const float* sub_rb2 = (const float*)d_in[9];
  const float* out_cw  = (const float*)d_in[10];
  const float* out_cb  = (const float*)d_in[11];
  const float* out_rw1 = (const float*)d_in[12];
  const float* out_rb1 = (const float*)d_in[13];
  const float* out_rg  = (const float*)d_in[14];
  const float* out_rbt = (const float*)d_in[15];
  const float* out_rw2 = (const float*)d_in[16];
  const float* out_rb2 = (const float*)d_in[17];

  char* base = (char*)d_ws;
  size_t off = 0;
  auto carve = [&](size_t bytes) -> char* {
    char* r = base + off;
    off += (bytes + 255) & ~(size_t)255;
    return r;
  };
  bf16*  nfsB    = (bf16*)carve((size_t)NIMG * CCH * HW * 2);
  bf16*  nfsT    = (bf16*)carve((size_t)NIMG * CCH * HW * 2);
  float* sivraw  = (float*)carve((size_t)2 * NIMG * CCH * 4);
  bf16*  sivbf   = (bf16*)carve((size_t)2 * NIMG * CCH * 2);
  float* corr    = (float*)carve((size_t)NIMG * HW * 32 * 4);
  float* csa     = (float*)carve((size_t)NIMG * HW * 4);
  int*   rank    = (int*)carve((size_t)NIMG * HW * 4);
  bf16*  cwp     = (bf16*)carve((size_t)NIMG * 128 * HW * 2);
  bf16*  pbf     = (bf16*)carve((size_t)NIMG * 128 * CCH * 2);
  float* xres    = (float*)carve((size_t)NIMG * 128 * HW * 4);
  float* zbuf    = (float*)carve((size_t)NIMG * 128 * HW * 4);
  bf16*  padT    = (bf16*)carve((size_t)NIMG * P2 * 128 * 2);
  bf16*  catT    = (bf16*)carve((size_t)NIMG * HW * 256 * 2);
  bf16*  wpkS1   = (bf16*)carve((size_t)9 * 128 * 128 * 2);
  bf16*  wpkS2   = (bf16*)carve((size_t)9 * 128 * 128 * 2);
  bf16*  wpkO1   = (bf16*)carve((size_t)9 * 128 * 128 * 2);
  bf16*  wpkO2   = (bf16*)carve((size_t)9 * 128 * 128 * 2);
  bf16*  owbf    = (bf16*)carve((size_t)128 * 256 * 2);
  float* bnscale = (float*)carve(128 * 4);
  float* bnshift = (float*)carve(128 * 4);

  auto res_block = [&](const float* xin, const bf16* wp1, const float* b1,
                       const float* g, const float* bt, const bf16* wp2,
                       const float* b2, int mode, float* outf, bf16* outb, int coff) {
    k_pad<<<dim3(P2, NIMG), 128, 0, stream>>>(xin, padT, nullptr, nullptr, 0);
    k_conv3<<<dim3(64, NIMG), 64, 0, stream>>>(wp1, padT, b1, nullptr, zbuf, nullptr, 0, 0);
    k_bn_stats<<<dim3(128), 256, 0, stream>>>(zbuf, g, bt, bnscale, bnshift);
    k_pad<<<dim3(P2, NIMG), 128, 0, stream>>>(zbuf, padT, bnscale, bnshift, 1);
    k_conv3<<<dim3(64, NIMG), 64, 0, stream>>>(wp2, padT, b2, xin, outf, outb, mode, coff);
  };

  // weight packing
  k_pack_convw<<<(9 * 128 * 128 + 255) / 256, 256, 0, stream>>>(sub_rw1, wpkS1);
  k_pack_convw<<<(9 * 128 * 128 + 255) / 256, 256, 0, stream>>>(sub_rw2, wpkS2);
  k_pack_convw<<<(9 * 128 * 128 + 255) / 256, 256, 0, stream>>>(out_rw1, wpkO1);
  k_pack_convw<<<(9 * 128 * 128 + 255) / 256, 256, 0, stream>>>(out_rw2, wpkO2);
  k_pack_ow<<<128, 256, 0, stream>>>(out_cw, owbf);

  // shared normalization + SIVs for both masks
  k_norm_feats<<<dim3(HW / 256, NIMG), 256, 0, stream>>>(feats, nfsB, nfsT);
  k_siv<<<dim3(CCH, NIMG), 128, 0, stream>>>(nfsB, sisms, sivraw);
  k_siv_norm<<<dim3(NIMG, 2), 256, 0, stream>>>(sivraw, sivbf);

  for (int mask = 0; mask < 2; ++mask) {
    const bf16* sivm = sivbf + (size_t)mask * NIMG * CCH;
    k_gemm_corr<<<dim3(16, NIMG), 128, 0, stream>>>(nfsT, sivm, corr);
    k_cm_norm<<<dim3(32, NIMG), 256, 0, stream>>>(corr);
    k_csa<<<NIMG, 256, 0, stream>>>(corr, csa);
    k_rank<<<NIMG, 256, 0, stream>>>(csa, rank);
    k_permw<<<dim3(128, NIMG), 256, 0, stream>>>(sub_cw, rank, cwp);
    k_gemm1<<<dim3(8, 2, NIMG), 128, 0, stream>>>(cwp, nfsB, pbf);
    k_gemm2<<<dim3(8, 4, NIMG), 128, 0, stream>>>(pbf, nfsT, csa, sub_cb, xres);
    res_block(xres, wpkS1, sub_rb1, sub_rg, sub_rbt, wpkS2, sub_rb2,
              /*mode=*/2, nullptr, catT, mask * 128);
  }

  // outer 1x1 conv + res block -> final output
  k_gemm_out1x1<<<dim3(8, 4, NIMG), 128, 0, stream>>>(owbf, catT, out_cb, xres);
  res_block(xres, wpkO1, out_rb1, out_rg, out_rbt, wpkO2, out_rb2,
            /*mode=*/1, (float*)d_out, nullptr, 0);
}
